// DKVMNModel_14834817040436
// MI455X (gfx1250) — matrix operand order repaired and checked by reading, not compiled, not run on hardware
//
#include <hip/hip_runtime.h>
#include <hip/hip_bf16.h>

// ---------------------------------------------------------------------------
// DKVMN forward on gfx1250 (MI455X).
//   1) prep_weights : erase_w/add_w f32[200x200] -> bf16[224x224] zero-padded
//   1b) prep_readw  : read_w f32[50x250] -> bf16[64x256] zero-padded
//   2) corr_softmax : w = softmax(q_e @ Mk^T)  f32 [204800 x 50]
//   3) gemm_erase_add: WMMA bf16 16x16x32, one wave = 16 rows x 32 cols,
//                      4 WMMAs per k-step (erase/add x 2 n-tiles), fused
//                      embedding gather + sigmoid/tanh epilogue
//   4) scan_step    : sequential DKVMN memory update, Mv column-in-registers,
//                     one block per batch element -> reads [204800 x 200]
//   5) gemm_fc      : WMMA bf16 GEMM h = tanh([reads,q_e] @ read_w^T),
//                     one wave = 16 rows x full N=64 -> h_ws [204800 x 64]
//   6) bce_kernel   : logit = h @ pred_w + b, masked BCE, preds/y to d_out,
//                     deterministic per-block partial sums
//   7) final_loss   : deterministic reduction of partials -> d_out[0]
// ---------------------------------------------------------------------------

#define R_TOTAL 204800   // BATCH*SEQ
#define BATCH   1024
#define SEQ     200
#define MEMN    50
#define KDIM    50
#define VDIM    200
#define FCN     50
#define KPAD    224      // VDIM padded to multiple of 32 (7 k-steps)
#define NPAD    224      // VDIM padded to multiple of 32 (7 n-tile PAIRS)
#define NPAIRS  7
#define MTILES  (R_TOTAL/16)   // 12800
#define FCK     256      // (VDIM+KDIM)=250 padded to 256 (8 k-steps)
#define FCNP    64       // FC=50 padded to 64 (4 n-tiles)

typedef __attribute__((ext_vector_type(16))) __bf16 v16bf;
typedef __attribute__((ext_vector_type(8)))  __bf16 v8bf;
typedef __attribute__((ext_vector_type(8)))  float  v8f;

// ---------------------------------------------------------------- kernel 1
__global__ void __launch_bounds__(256)
prep_weights(const float* __restrict__ ew, const float* __restrict__ aw,
             __bf16* __restrict__ wt_e, __bf16* __restrict__ wt_a)
{
    int i = blockIdx.x * 256 + threadIdx.x;          // over NPAD*KPAD
    if (i >= NPAD * KPAD) return;
    int n = i / KPAD, k = i % KPAD;
    float e = 0.f, a = 0.f;
    if (n < VDIM && k < VDIM) {
        e = ew[n * VDIM + k];
        a = aw[n * VDIM + k];
    }
    wt_e[i] = (__bf16)e;
    wt_a[i] = (__bf16)a;
}

// ---------------------------------------------------------------- kernel 1b
__global__ void __launch_bounds__(256)
prep_readw(const float* __restrict__ rw, __bf16* __restrict__ wt_r)
{
    int i = blockIdx.x * 256 + threadIdx.x;          // over FCNP*FCK
    if (i >= FCNP * FCK) return;
    int n = i / FCK, k = i % FCK;
    float v = 0.f;
    if (n < FCN && k < (VDIM + KDIM)) v = rw[n * (VDIM + KDIM) + k];
    wt_r[i] = (__bf16)v;
}

// ---------------------------------------------------------------- kernel 2
__global__ void __launch_bounds__(256)
corr_softmax(const int* __restrict__ q_data, const float* __restrict__ q_embed,
             const float* __restrict__ Mk, float* __restrict__ w_out)
{
    int row = blockIdx.x * 256 + threadIdx.x;
    if (row >= R_TOTAL) return;
    int idx = q_data[row];
    const float* q = q_embed + (size_t)idx * KDIM;
    float qr[KDIM];
#pragma unroll
    for (int k = 0; k < KDIM; ++k) qr[k] = q[k];
    float lg[MEMN];
    float mx = -1e30f;
#pragma unroll
    for (int m = 0; m < MEMN; ++m) {
        float acc = 0.f;
#pragma unroll
        for (int k = 0; k < KDIM; ++k) acc += qr[k] * Mk[m * KDIM + k];
        lg[m] = acc;
        mx = fmaxf(mx, acc);
    }
    float s = 0.f;
#pragma unroll
    for (int m = 0; m < MEMN; ++m) { float e = __expf(lg[m] - mx); lg[m] = e; s += e; }
    float inv = 1.f / s;
#pragma unroll
    for (int m = 0; m < MEMN; ++m) w_out[(size_t)row * MEMN + m] = lg[m] * inv;
}

// ------------------------------------------------------------- B frag load
__device__ __forceinline__ v16bf load_bfrag(const __bf16* base, int kb)
{
    // B layout (32x16 bf16): lane (half,n) holds 16 contiguous k at kb+half*16
    v16bf b;
    v8bf lo = *(const v8bf*)(base + kb);
    v8bf hi = *(const v8bf*)(base + kb + 8);
#pragma unroll
    for (int e = 0; e < 8; ++e) { b[e] = lo[e]; b[e + 8] = hi[e]; }
    return b;
}

// ---------------------------------------------------------------- kernel 3
// One wave per 16x32 output tile; A fragment feeds 4 WMMAs per k-step.
__global__ void __launch_bounds__(256)
gemm_erase_add(const int*   __restrict__ qa_data,
               const float* __restrict__ qa_embed,
               const __bf16* __restrict__ wt_e,
               const __bf16* __restrict__ wt_a,
               const float* __restrict__ erase_b,
               const float* __restrict__ add_b,
               float* __restrict__ erase_out,
               float* __restrict__ add_out)
{
    const int lane = threadIdx.x & 31;
    const int wid  = blockIdx.x * (blockDim.x >> 5) + (threadIdx.x >> 5);
    if (wid >= MTILES * NPAIRS) return;              // wave-uniform
    const int mt   = wid / NPAIRS;
    const int pr   = wid % NPAIRS;
    const int half = lane >> 4;
    const int l16  = lane & 15;

    const int    arow  = mt * 16 + l16;
    const int    aidx  = qa_data[arow];
    const float* abase = qa_embed + (size_t)aidx * VDIM;

    const int     nrow0 = (pr * 2 + 0) * 16 + l16;   // < NPAD
    const int     nrow1 = (pr * 2 + 1) * 16 + l16;
    const __bf16* be0 = wt_e + (size_t)nrow0 * KPAD;
    const __bf16* be1 = wt_e + (size_t)nrow1 * KPAD;
    const __bf16* ba0 = wt_a + (size_t)nrow0 * KPAD;
    const __bf16* ba1 = wt_a + (size_t)nrow1 * KPAD;

    v8f acc_e0 = {0.f,0.f,0.f,0.f,0.f,0.f,0.f,0.f};
    v8f acc_e1 = acc_e0, acc_a0 = acc_e0, acc_a1 = acc_e0;

#pragma unroll
    for (int kk = 0; kk < KPAD / 32; ++kk) {
        const int kbase = kk * 32;

        // A fragment: lanes 0-15 k=[kbase,+16); lanes 16-31 k=[kbase+16,+32)
        v16bf a;
        {
            float f[16];
            const int k0 = kbase + half * 16;
            const int k1 = kbase + half * 16 + 8;
            if (k0 + 8 <= VDIM) {
                const float4* p = (const float4*)(abase + k0);
                float4 u = p[0], v = p[1];
                f[0]=u.x; f[1]=u.y; f[2]=u.z; f[3]=u.w;
                f[4]=v.x; f[5]=v.y; f[6]=v.z; f[7]=v.w;
            } else {
#pragma unroll
                for (int e = 0; e < 8; ++e) f[e] = 0.f;
            }
            if (k1 + 8 <= VDIM) {
                const float4* p = (const float4*)(abase + k1);
                float4 u = p[0], v = p[1];
                f[8]=u.x;  f[9]=u.y;  f[10]=u.z; f[11]=u.w;
                f[12]=v.x; f[13]=v.y; f[14]=v.z; f[15]=v.w;
            } else {
#pragma unroll
                for (int e = 8; e < 16; ++e) f[e] = 0.f;
            }
#pragma unroll
            for (int e = 0; e < 16; ++e) a[e] = (__bf16)f[e];
        }

        const int kb = kbase + half * 16;
        v16bf fe0 = load_bfrag(be0, kb);
        v16bf fe1 = load_bfrag(be1, kb);
        v16bf fa0 = load_bfrag(ba0, kb);
        v16bf fa1 = load_bfrag(ba1, kb);

        acc_e0 = __builtin_amdgcn_wmma_f32_16x16x32_bf16(false, a, false, fe0, (short)0, acc_e0, false, false);
        acc_e1 = __builtin_amdgcn_wmma_f32_16x16x32_bf16(false, a, false, fe1, (short)0, acc_e1, false, false);
        acc_a0 = __builtin_amdgcn_wmma_f32_16x16x32_bf16(false, a, false, fa0, (short)0, acc_a0, false, false);
        acc_a1 = __builtin_amdgcn_wmma_f32_16x16x32_bf16(false, a, false, fa1, (short)0, acc_a1, false, false);
    }

    // Epilogue: D layout m = 16*mt + 8*half + j, n = ntile*16 + l16
#pragma unroll
    for (int t = 0; t < 2; ++t) {
        const int ncol = (pr * 2 + t) * 16 + l16;
        if (ncol < VDIM) {
            const float bbe = erase_b[ncol];
            const float bba = add_b[ncol];
            const v8f* pe = t ? &acc_e1 : &acc_e0;
            const v8f* pa = t ? &acc_a1 : &acc_a0;
#pragma unroll
            for (int j = 0; j < 8; ++j) {
                const int m  = mt * 16 + half * 8 + j;
                const float xe = (*pe)[j] + bbe;
                const float xa = (*pa)[j] + bba;
                erase_out[(size_t)m * VDIM + ncol] = 1.f / (1.f + __expf(-xe));
                add_out  [(size_t)m * VDIM + ncol] = tanhf(xa);
            }
        }
    }
}

// ---------------------------------------------------------------- kernel 4
// One block per batch element. Thread v owns Mv[:, v] in 50 registers.
__global__ void __launch_bounds__(256)
scan_step(const float* __restrict__ Mv0,
          const float* __restrict__ w_ws,
          const float* __restrict__ erase_ws,
          const float* __restrict__ add_ws,
          float* __restrict__ reads_ws)
{
    const int b   = blockIdx.x;
    const int tid = threadIdx.x;
    __shared__ float w_s[64];

    float mv[MEMN];
    if (tid < VDIM) {
#pragma unroll
        for (int m = 0; m < MEMN; ++m) mv[m] = Mv0[m * VDIM + tid];
    }

    for (int s = 0; s < SEQ; ++s) {
        const size_t row = (size_t)b * SEQ + s;
        if (tid < MEMN) w_s[tid] = w_ws[row * MEMN + tid];
        __syncthreads();
        if (tid < VDIM) {
            const float e = erase_ws[row * VDIM + tid];
            const float a = add_ws  [row * VDIM + tid];
            float acc = 0.f;
#pragma unroll
            for (int m = 0; m < MEMN; ++m) {
                const float wm = w_s[m];
                const float v  = mv[m];
                acc += wm * v;
                mv[m] = v * (1.f - wm * e) + wm * a;
            }
            reads_ws[row * VDIM + tid] = acc;
        }
        __syncthreads();
    }
}

// ---------------------------------------------------------------- kernel 5
// FC layer h = tanh([reads | q_e] @ read_w^T + read_b) via WMMA.
// One wave per 16-row tile, full N=64 (4 accumulators), A loaded once.
__global__ void __launch_bounds__(256)
gemm_fc(const int*   __restrict__ q_data,
        const float* __restrict__ q_embed,
        const float* __restrict__ reads_ws,
        const __bf16* __restrict__ wt_r,
        const float* __restrict__ read_b,
        float* __restrict__ h_ws)            // [R_TOTAL x FCNP]
{
    const int lane = threadIdx.x & 31;
    const int mt   = blockIdx.x * (blockDim.x >> 5) + (threadIdx.x >> 5);
    if (mt >= MTILES) return;                        // wave-uniform
    const int half = lane >> 4;
    const int l16  = lane & 15;

    const int    arow = mt * 16 + l16;
    const float* rrow = reads_ws + (size_t)arow * VDIM;
    const float* qrow = q_embed + (size_t)q_data[arow] * KDIM;  // 8B-aligned only

    v8f acc[4];
#pragma unroll
    for (int t = 0; t < 4; ++t) acc[t] = (v8f){0.f,0.f,0.f,0.f,0.f,0.f,0.f,0.f};

#pragma unroll
    for (int kk = 0; kk < FCK / 32; ++kk) {
        const int kbase = kk * 32;

        v16bf a;
        {
            float f[16];
#pragma unroll
            for (int c = 0; c < 2; ++c) {
                const int k = kbase + half * 16 + c * 8;   // chunk start, mult of 8
                float* fd = f + c * 8;
                if (k + 8 <= VDIM) {                       // reads region
                    const float4* p = (const float4*)(rrow + k);
                    float4 u = p[0], v = p[1];
                    fd[0]=u.x; fd[1]=u.y; fd[2]=u.z; fd[3]=u.w;
                    fd[4]=v.x; fd[5]=v.y; fd[6]=v.z; fd[7]=v.w;
                } else if (k >= VDIM && k + 8 <= VDIM + KDIM) {  // q_e region (float2: 8B align)
                    const float2* p = (const float2*)(qrow + (k - VDIM));
                    float2 a0 = p[0], a1 = p[1], a2 = p[2], a3 = p[3];
                    fd[0]=a0.x; fd[1]=a0.y; fd[2]=a1.x; fd[3]=a1.y;
                    fd[4]=a2.x; fd[5]=a2.y; fd[6]=a3.x; fd[7]=a3.y;
                } else {                                   // mixed / pad chunk
#pragma unroll
                    for (int e = 0; e < 8; ++e) {
                        const int kk2 = k + e;
                        fd[e] = (kk2 >= VDIM && kk2 < VDIM + KDIM) ? qrow[kk2 - VDIM] : 0.f;
                    }
                }
            }
#pragma unroll
            for (int e = 0; e < 16; ++e) a[e] = (__bf16)f[e];
        }

        const int kb = kbase + half * 16;
#pragma unroll
        for (int t = 0; t < 4; ++t) {
            const __bf16* bbase = wt_r + (size_t)(t * 16 + l16) * FCK;
            v16bf bf = load_bfrag(bbase, kb);
            acc[t] = __builtin_amdgcn_wmma_f32_16x16x32_bf16(false, a, false, bf, (short)0, acc[t], false, false);
        }
    }

    // Epilogue: tanh(x + bias); pad cols (>=FCN) get bias 0 -> never read later
#pragma unroll
    for (int t = 0; t < 4; ++t) {
        const int ncol = t * 16 + l16;                 // < 64
        const float bb = (ncol < FCN) ? read_b[ncol] : 0.f;
#pragma unroll
        for (int j = 0; j < 8; ++j) {
            const int m = mt * 16 + half * 8 + j;
            h_ws[(size_t)m * FCNP + ncol] = tanhf(acc[t][j] + bb);
        }
    }
}

// ---------------------------------------------------------------- kernel 6
__global__ void __launch_bounds__(256)
bce_kernel(const int*   __restrict__ target,
           const float* __restrict__ h_ws,
           const float* __restrict__ pred_w,
           const float* __restrict__ pred_b,
           float* __restrict__ out,            // [0]=loss [1..]=preds [..]=y
           float* __restrict__ partials,
           float* __restrict__ pcounts)
{
    const int row = blockIdx.x * 256 + threadIdx.x;
    float bce = 0.f, cnt = 0.f;
    if (row < R_TOTAL) {
        const float* hr = h_ws + (size_t)row * FCNP;
        float logit = pred_b[0];
#pragma unroll
        for (int j = 0; j < FCN; ++j) logit += hr[j] * pred_w[j];

        const int  t    = target[row];
        const bool mask = (t >= 0);
        const float y   = (float)t;
        const float l   = logit;
        const float b_  = fmaxf(l, 0.f) - l * y + log1pf(__expf(-fabsf(l)));
        bce = mask ? b_ : 0.f;
        cnt = mask ? 1.f : 0.f;
        out[1 + row]           = mask ? 1.f / (1.f + __expf(-l)) : 0.f;
        out[1 + R_TOTAL + row] = mask ? y : 0.f;
    }
    __shared__ float r1[256], r2[256];
    r1[threadIdx.x] = bce;
    r2[threadIdx.x] = cnt;
    __syncthreads();
    for (int o = 128; o > 0; o >>= 1) {
        if (threadIdx.x < o) {
            r1[threadIdx.x] += r1[threadIdx.x + o];
            r2[threadIdx.x] += r2[threadIdx.x + o];
        }
        __syncthreads();
    }
    if (threadIdx.x == 0) {
        partials[blockIdx.x] = r1[0];
        pcounts [blockIdx.x] = r2[0];
    }
}

// ---------------------------------------------------------------- kernel 7
__global__ void __launch_bounds__(256)
final_loss(const float* __restrict__ partials, const float* __restrict__ pcounts,
           int n, float* __restrict__ out)
{
    __shared__ float r1[256], r2[256];
    float s1 = 0.f, s2 = 0.f;
    for (int i = threadIdx.x; i < n; i += 256) { s1 += partials[i]; s2 += pcounts[i]; }
    r1[threadIdx.x] = s1;
    r2[threadIdx.x] = s2;
    __syncthreads();
    for (int o = 128; o > 0; o >>= 1) {
        if (threadIdx.x < o) {
            r1[threadIdx.x] += r1[threadIdx.x + o];
            r2[threadIdx.x] += r2[threadIdx.x + o];
        }
        __syncthreads();
    }
    if (threadIdx.x == 0) out[0] = r1[0] / fmaxf(r2[0], 1.0f);
}

// ---------------------------------------------------------------- launch
extern "C" void kernel_launch(void* const* d_in, const int* in_sizes, int n_in,
                              void* d_out, int out_size, void* d_ws, size_t ws_size,
                              hipStream_t stream)
{
    const int*   q_data   = (const int*)  d_in[0];
    const int*   qa_data  = (const int*)  d_in[1];
    const int*   target   = (const int*)  d_in[2];
    const float* q_embed  = (const float*)d_in[3];
    const float* qa_embed = (const float*)d_in[4];
    const float* Mk       = (const float*)d_in[5];
    const float* Mv0      = (const float*)d_in[6];
    const float* erase_w  = (const float*)d_in[7];
    const float* erase_b  = (const float*)d_in[8];
    const float* add_w    = (const float*)d_in[9];
    const float* add_b    = (const float*)d_in[10];
    const float* read_w   = (const float*)d_in[11];
    const float* read_b   = (const float*)d_in[12];
    const float* pred_w   = (const float*)d_in[13];
    const float* pred_b   = (const float*)d_in[14];
    float* out = (float*)d_out;

    // workspace carve-up (~587 MB total), 256B aligned regions
    char*  ws  = (char*)d_ws;
    size_t off = 0;
    auto carve = [&](size_t bytes) -> void* {
        void* p = ws + off;
        off = (off + bytes + 255) & ~(size_t)255;
        return p;
    };
    float*  w_ws     = (float*) carve((size_t)R_TOTAL * MEMN * 4);
    float*  erase_ws = (float*) carve((size_t)R_TOTAL * VDIM * 4);
    float*  add_ws   = (float*) carve((size_t)R_TOTAL * VDIM * 4);
    float*  reads_ws = (float*) carve((size_t)R_TOTAL * VDIM * 4);
    float*  h_ws     = (float*) carve((size_t)R_TOTAL * FCNP * 4);
    __bf16* wt_e     = (__bf16*)carve((size_t)NPAD * KPAD * 2);
    __bf16* wt_a     = (__bf16*)carve((size_t)NPAD * KPAD * 2);
    __bf16* wt_r     = (__bf16*)carve((size_t)FCNP * FCK * 2);
    float*  partials = (float*) carve(800 * 4);
    float*  pcounts  = (float*) carve(800 * 4);
    (void)ws_size; (void)in_sizes; (void)n_in; (void)out_size;

    prep_weights  <<<(NPAD * KPAD + 255) / 256, 256, 0, stream>>>(erase_w, add_w, wt_e, wt_a);
    prep_readw    <<<(FCNP * FCK + 255) / 256, 256, 0, stream>>>(read_w, wt_r);
    corr_softmax  <<<R_TOTAL / 256, 256, 0, stream>>>(q_data, q_embed, Mk, w_ws);
    gemm_erase_add<<<(MTILES * NPAIRS) / 8, 256, 0, stream>>>(qa_data, qa_embed, wt_e, wt_a,
                                                              erase_b, add_b, erase_ws, add_ws);
    scan_step     <<<BATCH, 256, 0, stream>>>(Mv0, w_ws, erase_ws, add_ws, reads_ws);
    gemm_fc       <<<MTILES / 8, 256, 0, stream>>>(q_data, q_embed, reads_ws, wt_r, read_b, h_ws);
    bce_kernel    <<<R_TOTAL / 256, 256, 0, stream>>>(target, h_ws, pred_w, pred_b,
                                                      out, partials, pcounts);
    final_loss    <<<1, 256, 0, stream>>>(partials, pcounts, R_TOTAL / 256, out);
}
